// GraphGPSRegressor_36507222016093
// MI455X (gfx1250) — compile-verified, matrix operand order checked by reading, not run on hardware
//
#include <hip/hip_runtime.h>
#include <hip/hip_bf16.h>
#include <math.h>

// ---------------------------------------------------------------------------
// GraphGPS forward for MI455X (gfx1250, wave32, WMMA bf16), round 3:
// - GEMM unchanged (verified clean b128 -> v_wmma pipeline)
// - attention: 64-key steps, block-shared K/V tiles staged into LDS with
//   GLOBAL_LOAD_ASYNC_TO_LDS_B128 (ASYNCcnt) + s_wait_asynccnt
// ---------------------------------------------------------------------------

typedef __attribute__((ext_vector_type(16))) __bf16 v16bf;
typedef __attribute__((ext_vector_type(8)))  __bf16 v8bf;
typedef __attribute__((ext_vector_type(4)))  __bf16 v4bf;
typedef __attribute__((ext_vector_type(8)))  float  v8f;

#define NN    4096
#define EE    131072
#define DIN   72
#define DINP  96      // DIN padded to K multiple of 32
#define DD    256
#define HH    8
#define HDIM  32
#define LLAY  2
#define DEMB  128
#define TOUT  5
#define TOUTP 64      // head output rows padded to one 64-wide N tile
#define FF    512
#define LN_EPS 1e-5f

#define BM 128
#define BN 64

// flip to disable the async global->LDS path if the assembler rejects it
//#define NO_ASYNC_LDS 1

__device__ __forceinline__ float act_apply(float x, int act) {
  if (act == 1) {                       // ELU
    return x > 0.f ? x : __expf(x) - 1.f;
  }
  if (act == 2) {                       // GELU (tanh approx, jax default)
    const float c = 0.7978845608028654f;
    float t = tanhf(c * (x + 0.044715f * x * x * x));
    return 0.5f * x * (1.f + t);
  }
  return x;
}

// Load one 16-bit A/B fragment: two 16-byte runs at [ksel, +8) and [16+ksel, +8).
__device__ __forceinline__ v16bf load_frag(const __bf16* __restrict__ p, int ksel) {
  v8bf lo = *(const v8bf*)(p + ksel);
  v8bf hi = *(const v8bf*)(p + 16 + ksel);
  return __builtin_shufflevector(lo, hi, 0, 1, 2, 3, 4, 5, 6, 7,
                                 8, 9, 10, 11, 12, 13, 14, 15);
}

// 16B global -> LDS copy; async DMA variant tracked by ASYNCcnt.
__device__ __forceinline__ void stage16(__bf16* lds_dst, const __bf16* gsrc) {
#ifndef NO_ASYNC_LDS
  unsigned int  loff  = (unsigned int)(size_t)lds_dst;
  unsigned long long gaddr = (unsigned long long)(size_t)gsrc;
  asm volatile("global_load_async_to_lds_b128 %0, %1, off"
               :: "v"(loff), "v"(gaddr) : "memory");
#else
  *(v8bf*)lds_dst = *(const v8bf*)gsrc;
#endif
}

__device__ __forceinline__ void stage_wait() {
#ifndef NO_ASYNC_LDS
  asm volatile("s_wait_asynccnt 0x0" ::: "memory");
#endif
}

// ---------------------------------------------------------------------------
// WMMA GEMM: out[M,Nn] = act(A[M,K]*W[Nn,K]^T + bias).  A,W bf16 (K%32==0,
// W rows padded to 64*gridY). Block=256 thr (8 waves), tile 128x64, no LDS.
// ---------------------------------------------------------------------------
__global__ __launch_bounds__(256)
void wmma_gemm_bf16(const __bf16* __restrict__ A, const __bf16* __restrict__ W,
                    const float* __restrict__ bias, float* __restrict__ outF,
                    __bf16* __restrict__ outB, int M, int Nn, int K, int act)
{
  const int lane   = threadIdx.x & 31;
  const int wid    = threadIdx.x >> 5;
  const int laneHi = lane >> 4;
  const int lanelo = lane & 15;
  const int ksel   = laneHi * 8;
  const int m0     = blockIdx.x * BM;
  const int n0     = blockIdx.y * BN;

  const __bf16* arow = A + (size_t)(m0 + wid * 16 + lanelo) * K;
  const __bf16* brow = W + (size_t)(n0 + lanelo) * K;

  v8f acc[4] = {};

#pragma unroll 2
  for (int k0 = 0; k0 < K; k0 += 32) {
    v16bf af = load_frag(arow + k0, ksel);
#pragma unroll
    for (int nt = 0; nt < 4; ++nt) {
      v16bf bfv = load_frag(brow + (size_t)nt * 16 * K + k0, ksel);
      acc[nt] = __builtin_amdgcn_wmma_f32_16x16x32_bf16(
          false, af, false, bfv, (short)0, acc[nt], false, false);
    }
  }

#pragma unroll
  for (int nt = 0; nt < 4; ++nt) {
    int col = n0 + nt * 16 + lanelo;
    if (col < Nn) {
      float bv = bias[col];
#pragma unroll
      for (int i = 0; i < 8; ++i) {
        int row = m0 + wid * 16 + i + 8 * laneHi;
        float val = act_apply(acc[nt][i] + bv, act);
        size_t off = (size_t)row * Nn + col;
        if (outF) outF[off] = val;
        if (outB) outB[off] = (__bf16)val;
      }
    }
  }
}

// ---------------------------------------------------------------------------
// Flash attention, 64-key steps with LDS-staged shared K/V tiles.
// grid=(N/128, H), block=256 (8 waves); each wave owns 16 query rows.
// Qb/Kb: [H][N][32] bf16. Vt: [H][32][N] bf16. Output bf16 [N][256].
// ---------------------------------------------------------------------------
#define KSTR 40   // K-tile row stride (80B: aligned + conflict-free)
#define VSTR 72   // V-tile / P-buf row stride (144B: aligned + conflict-free)

__global__ __launch_bounds__(256)
void attn_flash(const __bf16* __restrict__ Qb, const __bf16* __restrict__ Kb,
                const __bf16* __restrict__ Vt, __bf16* __restrict__ Ob,
                float scale)
{
  __shared__ __bf16 Ktile[64 * KSTR];        // 64 keys x 32 dims
  __shared__ __bf16 Vtile[32 * VSTR];        // 32 dims x 64 keys
  __shared__ __bf16 pbuf[8][16 * VSTR];      // per-wave 16 x 64 P tile

  const int t      = threadIdx.x;
  const int lane   = t & 31;
  const int wid    = t >> 5;
  const int laneHi = lane >> 4;
  const int lanelo = lane & 15;
  const int ksel   = laneHi * 8;
  const int h      = blockIdx.y;
  const int q0     = blockIdx.x * 128 + wid * 16;

  // staging assignments (one 16B chunk per thread per tile)
  const int kr = t >> 2, kc = (t & 3) * 8;   // K tile: 64 rows x 4 chunks
  const int vr = t >> 3, vc = (t & 7) * 8;   // V tile: 32 rows x 8 chunks
  const __bf16* ksrc = Kb + ((size_t)h * NN + kr) * HDIM + kc;
  const __bf16* vsrc = Vt + ((size_t)h * HDIM + vr) * NN + vc;
  __bf16* kdst = &Ktile[kr * KSTR + kc];
  __bf16* vdst = &Vtile[vr * VSTR + vc];

  v16bf qf = load_frag(Qb + ((size_t)h * NN + (q0 + lanelo)) * HDIM, ksel);

  v8f o0 = {}, o1 = {};
  float mrow[8], lrow[8];
#pragma unroll
  for (int i = 0; i < 8; ++i) { mrow[i] = -3.0e38f; lrow[i] = 0.f; }

  __bf16* pw = &pbuf[wid][0];

  for (int kv = 0; kv < NN; kv += 64) {
    __syncthreads();                          // previous tile reads complete
    stage16(kdst, ksrc + (size_t)kv * HDIM);  // K rows advance by kv
    stage16(vdst, vsrc + kv);                 // V cols advance by kv
    stage_wait();
    __syncthreads();                          // tiles visible to all waves

    // S = Q * K^T for 4 sub-tiles of 16 keys
    v8f z = {};
    v8f s[4];
#pragma unroll
    for (int st = 0; st < 4; ++st) {
      v16bf kf = load_frag(&Ktile[(st * 16 + lanelo) * KSTR], ksel);
      s[st] = __builtin_amdgcn_wmma_f32_16x16x32_bf16(false, qf, false, kf,
                                                      (short)0, z, false, false);
    }

    // online softmax over 64 keys (stats once per 64)
#pragma unroll
    for (int i = 0; i < 8; ++i) {
      float a0 = s[0][i] * scale, a1 = s[1][i] * scale;
      float a2 = s[2][i] * scale, a3 = s[3][i] * scale;
      float tmax = fmaxf(fmaxf(a0, a1), fmaxf(a2, a3));
      tmax = fmaxf(tmax, __shfl_xor(tmax, 1, 16));
      tmax = fmaxf(tmax, __shfl_xor(tmax, 2, 16));
      tmax = fmaxf(tmax, __shfl_xor(tmax, 4, 16));
      tmax = fmaxf(tmax, __shfl_xor(tmax, 8, 16));
      float newm  = fmaxf(mrow[i], tmax);
      float alpha = __expf(mrow[i] - newm);
      mrow[i] = newm;
      float p0 = __expf(a0 - newm), p1 = __expf(a1 - newm);
      float p2 = __expf(a2 - newm), p3 = __expf(a3 - newm);
      float rs = (p0 + p1) + (p2 + p3);
      rs += __shfl_xor(rs, 1, 16);
      rs += __shfl_xor(rs, 2, 16);
      rs += __shfl_xor(rs, 4, 16);
      rs += __shfl_xor(rs, 8, 16);
      lrow[i] = lrow[i] * alpha + rs;
      o0[i] *= alpha; o1[i] *= alpha;
      int prow = (i + 8 * laneHi) * VSTR + lanelo;
      pw[prow]      = (__bf16)p0;
      pw[prow + 16] = (__bf16)p1;
      pw[prow + 32] = (__bf16)p2;
      pw[prow + 48] = (__bf16)p3;
    }

    // wave-private LDS round-trip: C-layout P -> two A fragments (32 keys ea)
    asm volatile("s_wait_dscnt 0x0" ::: "memory");
    v16bf pf0 = load_frag(&pw[lanelo * VSTR], ksel);
    v16bf pf1 = load_frag(&pw[lanelo * VSTR + 32], ksel);

    // V B-fragments: [dim half][key chunk]
    v16bf v00 = load_frag(&Vtile[lanelo * VSTR], ksel);
    v16bf v01 = load_frag(&Vtile[lanelo * VSTR + 32], ksel);
    v16bf v10 = load_frag(&Vtile[(16 + lanelo) * VSTR], ksel);
    v16bf v11 = load_frag(&Vtile[(16 + lanelo) * VSTR + 32], ksel);

    o0 = __builtin_amdgcn_wmma_f32_16x16x32_bf16(false, pf0, false, v00, (short)0, o0, false, false);
    o0 = __builtin_amdgcn_wmma_f32_16x16x32_bf16(false, pf1, false, v01, (short)0, o0, false, false);
    o1 = __builtin_amdgcn_wmma_f32_16x16x32_bf16(false, pf0, false, v10, (short)0, o1, false, false);
    o1 = __builtin_amdgcn_wmma_f32_16x16x32_bf16(false, pf1, false, v11, (short)0, o1, false, false);
  }

#pragma unroll
  for (int i = 0; i < 8; ++i) {
    int row = q0 + i + 8 * laneHi;
    float inv = 1.f / lrow[i];
    int colb = h * HDIM + lanelo;
    Ob[(size_t)row * DD + colb]      = (__bf16)(o0[i] * inv);
    Ob[(size_t)row * DD + colb + 16] = (__bf16)(o1[i] * inv);
  }
}

// ---------------------------------------------------------------------------
// fp32 -> bf16 with optional row/col padding (each weight, once per launch)
// ---------------------------------------------------------------------------
__global__ void convert_pad(const float* __restrict__ src, __bf16* __restrict__ dst,
                            int rows_out, int cols_out, int rows_in, int cols_in)
{
  int i = blockIdx.x * 256 + threadIdx.x;
  if (i >= rows_out * cols_out) return;
  int r = i / cols_out, c = i - r * cols_out;
  float v = (r < rows_in && c < cols_in) ? src[(size_t)r * cols_in + c] : 0.f;
  dst[i] = (__bf16)v;
}

// qkv bf16 [N][768] -> Qb/Kb [H][N][32], Vt [H][32][N]
__global__ void qkv_split(const __bf16* __restrict__ qkv, __bf16* __restrict__ Qb,
                          __bf16* __restrict__ Kb, __bf16* __restrict__ Vt)
{
  int idx = blockIdx.x * 256 + threadIdx.x;
  if (idx >= NN * DD) return;
  int n = idx >> 8;
  int c = idx & 255;
  int h = c >> 5, d = c & 31;
  const __bf16* row = qkv + (size_t)n * (3 * DD);
  Qb[((size_t)h * NN + n) * HDIM + d] = row[c];
  Kb[((size_t)h * NN + n) * HDIM + d] = row[DD + c];
  Vt[((size_t)h * HDIM + d) * NN + n] = row[2 * DD + c];
}

// GIN scatter-add: 8 threads/edge, 32 dims each (global_atomic_add_f32)
__global__ void edge_scatter(const long long* __restrict__ ei,
                             const float* __restrict__ X, float* __restrict__ AGG)
{
  int tid = blockIdx.x * 256 + threadIdx.x;
  int e = tid >> 3;
  int ch = tid & 7;
  if (e >= EE) return;
  int s = (int)ei[e];
  int d = (int)ei[EE + e];
  const float* xs = X + (size_t)s * DD + ch * 32;
  float* ad = AGG + (size_t)d * DD + ch * 32;
#pragma unroll
  for (int i = 0; i < 32; ++i) atomicAdd(&ad[i], xs[i]);
}

__global__ void fill_zero4(float4* __restrict__ p, int n4) {
  int i = blockIdx.x * 256 + threadIdx.x;
  if (i < n4) p[i] = make_float4(0.f, 0.f, 0.f, 0.f);
}

// outB = bf16((1+eps)*X + AGG), 4 elems/thread
__global__ void gin_combine(const float4* __restrict__ X, const float4* __restrict__ AGG,
                            const float* __restrict__ eps, v4bf* __restrict__ outB, int n4)
{
  int i = blockIdx.x * 256 + threadIdx.x;
  if (i >= n4) return;
  float e = 1.f + eps[0];
  float4 x = X[i], a = AGG[i];
  v4bf o;
  o[0] = (__bf16)(e * x.x + a.x);
  o[1] = (__bf16)(e * x.y + a.y);
  o[2] = (__bf16)(e * x.z + a.z);
  o[3] = (__bf16)(e * x.w + a.w);
  outB[i] = o;
}

// out / outBf = res + act(LN(in)*g + b); one wave per 256-wide row.
__global__ __launch_bounds__(256)
void ln_res(const float* __restrict__ in, const float* __restrict__ g,
            const float* __restrict__ b, const float* __restrict__ res,
            float* __restrict__ out, __bf16* __restrict__ outBf, int act)
{
  int lane = threadIdx.x & 31;
  int wid  = threadIdx.x >> 5;
  int row  = blockIdx.x * 8 + wid;
  const float* rp = in + (size_t)row * DD + lane * 8;
  float v[8];
  float s = 0.f, s2 = 0.f;
#pragma unroll
  for (int i = 0; i < 8; ++i) {
    float x = rp[i];
    v[i] = x; s += x; s2 += x * x;
  }
#pragma unroll
  for (int m = 1; m < 32; m <<= 1) {
    s  += __shfl_xor(s,  m, 32);
    s2 += __shfl_xor(s2, m, 32);
  }
  float mu   = s * (1.f / DD);
  float var  = s2 * (1.f / DD) - mu * mu;
  float rstd = rsqrtf(var + LN_EPS);
  size_t base = (size_t)row * DD + lane * 8;
#pragma unroll
  for (int i = 0; i < 8; ++i) {
    int c = lane * 8 + i;
    float y = (v[i] - mu) * rstd * g[c] + b[c];
    y = act_apply(y, act);
    float o = res[base + i] + y;
    out[base + i] = o;
    outBf[base + i] = (__bf16)o;
  }
}

// ---------------------------------------------------------------------------
// Orchestration
// ---------------------------------------------------------------------------
static inline void launch_gemm(const __bf16* A, const __bf16* W, const float* bias,
                               float* outF, __bf16* outB, int M, int Nc, int K,
                               int act, hipStream_t stream) {
  dim3 g((M + BM - 1) / BM, (Nc + BN - 1) / BN);
  wmma_gemm_bf16<<<g, 256, 0, stream>>>(A, W, bias, outF, outB, M, Nc, K, act);
}

static inline void launch_conv(const float* src, __bf16* dst, int ro, int co,
                               int ri, int ci, hipStream_t stream) {
  int n = ro * co;
  convert_pad<<<(n + 255) / 256, 256, 0, stream>>>(src, dst, ro, co, ri, ci);
}

extern "C" void kernel_launch(void* const* d_in, const int* in_sizes, int n_in,
                              void* d_out, int out_size, void* d_ws, size_t ws_size,
                              hipStream_t stream) {
  const float*     x_in       = (const float*)d_in[0];
  const long long* ei         = (const long long*)d_in[1];
  const float*     in_w       = (const float*)d_in[2];
  const float*     in_b       = (const float*)d_in[3];
  const float*     head_w1    = (const float*)d_in[4];
  const float*     head_b1    = (const float*)d_in[5];
  const float*     head_w2    = (const float*)d_in[6];
  const float*     head_b2    = (const float*)d_in[7];
  const float*     mlp_w1     = (const float*)d_in[8];
  const float*     mlp_b1     = (const float*)d_in[9];
  const float*     mlp_w2     = (const float*)d_in[10];
  const float*     mlp_b2     = (const float*)d_in[11];
  const float*     gin_eps    = (const float*)d_in[12];
  const float*     ln1_g      = (const float*)d_in[13];
  const float*     ln1_b      = (const float*)d_in[14];
  const float*     attn_in_w  = (const float*)d_in[15];
  const float*     attn_in_b  = (const float*)d_in[16];
  const float*     attn_out_w = (const float*)d_in[17];
  const float*     attn_out_b = (const float*)d_in[18];
  const float*     ln2_g      = (const float*)d_in[19];
  const float*     ln2_b      = (const float*)d_in[20];
  const float*     ffn_w1     = (const float*)d_in[21];
  const float*     ffn_b1     = (const float*)d_in[22];
  const float*     ffn_w2     = (const float*)d_in[23];
  const float*     ffn_b2     = (const float*)d_in[24];
  const float*     ln3_g      = (const float*)d_in[25];
  const float*     ln3_b      = (const float*)d_in[26];
  float* out = (float*)d_out;

  // ---- workspace carve ----
  float* X   = (float*)d_ws;                         // [N,256] residual stream
  float* AGG = X   + (size_t)NN * DD;                // [N,256]
  float* TF  = AGG + (size_t)NN * DD;                // [N,256] LN input scratch
  __bf16* bp = (__bf16*)(TF + (size_t)NN * DD);

  auto carve = [&](size_t n) { __bf16* p = bp; bp += n; return p; };
  __bf16* Xb  = carve((size_t)NN * DD);              // bf16 shadow of X
  __bf16* B1  = carve((size_t)NN * 3 * DD);          // gemm bf16 scratch (<=768 w)
  __bf16* B2  = carve((size_t)NN * DD);              // combine out / attn out
  __bf16* Qb  = carve((size_t)HH * NN * HDIM);
  __bf16* Kb  = carve((size_t)HH * NN * HDIM);
  __bf16* Vt  = carve((size_t)HH * NN * HDIM);
  __bf16* XPb = carve((size_t)NN * DINP);            // padded input x
  __bf16* HNb = carve((size_t)NN * DEMB);            // head hidden
  __bf16* Winb  = carve((size_t)DD * DINP);
  __bf16* Wm1b  = carve((size_t)LLAY * DD * DD);
  __bf16* Wm2b  = carve((size_t)LLAY * DD * DD);
  __bf16* Waib  = carve((size_t)LLAY * 3 * DD * DD);
  __bf16* Waob  = carve((size_t)LLAY * DD * DD);
  __bf16* Wf1b  = carve((size_t)LLAY * FF * DD);
  __bf16* Wf2b  = carve((size_t)LLAY * DD * FF);
  __bf16* Wh1b  = carve((size_t)DEMB * DD);
  __bf16* Wh2b  = carve((size_t)TOUTP * DEMB);

  const int nD = NN * DD;
  const float scale = 0.17677669529663687f;          // 1/sqrt(32)

  // ---- per-launch weight/input conversions to bf16 ----
  launch_conv(x_in, XPb, NN, DINP, NN, DIN, stream);
  launch_conv(in_w, Winb, DD, DINP, DD, DIN, stream);
  launch_conv(mlp_w1, Wm1b, LLAY * DD, DD, LLAY * DD, DD, stream);
  launch_conv(mlp_w2, Wm2b, LLAY * DD, DD, LLAY * DD, DD, stream);
  launch_conv(attn_in_w, Waib, LLAY * 3 * DD, DD, LLAY * 3 * DD, DD, stream);
  launch_conv(attn_out_w, Waob, LLAY * DD, DD, LLAY * DD, DD, stream);
  launch_conv(ffn_w1, Wf1b, LLAY * FF, DD, LLAY * FF, DD, stream);
  launch_conv(ffn_w2, Wf2b, LLAY * DD, FF, LLAY * DD, FF, stream);
  launch_conv(head_w1, Wh1b, DEMB, DD, DEMB, DD, stream);
  launch_conv(head_w2, Wh2b, TOUTP, DEMB, TOUT, DEMB, stream);

  // ---- input projection + ELU ----
  launch_gemm(XPb, Winb, in_b, X, Xb, NN, DD, DINP, 1, stream);

  for (int l = 0; l < LLAY; ++l) {
    // ---- GIN ----
    fill_zero4<<<(nD / 4 + 255) / 256, 256, 0, stream>>>((float4*)AGG, nD / 4);
    edge_scatter<<<(EE * 8 + 255) / 256, 256, 0, stream>>>(ei, X, AGG);
    gin_combine<<<(nD / 4 + 255) / 256, 256, 0, stream>>>(
        (const float4*)X, (const float4*)AGG, gin_eps + l, (v4bf*)B2, nD / 4);
    launch_gemm(B2, Wm1b + (size_t)l * DD * DD, mlp_b1 + l * DD, nullptr, B1,
                NN, DD, DD, 1, stream);
    launch_gemm(B1, Wm2b + (size_t)l * DD * DD, mlp_b2 + l * DD, TF, nullptr,
                NN, DD, DD, 0, stream);
    ln_res<<<NN / 8, 256, 0, stream>>>(TF, ln1_g + l * DD, ln1_b + l * DD, X, X, Xb, 1);

    // ---- global multi-head attention ----
    launch_gemm(Xb, Waib + (size_t)l * 3 * DD * DD, attn_in_b + l * 3 * DD,
                nullptr, B1, NN, 3 * DD, DD, 0, stream);
    qkv_split<<<(nD + 255) / 256, 256, 0, stream>>>(B1, Qb, Kb, Vt);
    dim3 ag(NN / 128, HH);
    attn_flash<<<ag, 256, 0, stream>>>(Qb, Kb, Vt, B2, scale);
    launch_gemm(B2, Waob + (size_t)l * DD * DD, attn_out_b + l * DD, TF, nullptr,
                NN, DD, DD, 0, stream);
    ln_res<<<NN / 8, 256, 0, stream>>>(TF, ln2_g + l * DD, ln2_b + l * DD, X, X, Xb, 0);

    // ---- FFN ----
    launch_gemm(Xb, Wf1b + (size_t)l * FF * DD, ffn_b1 + l * FF, nullptr, B1,
                NN, FF, DD, 2, stream);
    launch_gemm(B1, Wf2b + (size_t)l * DD * FF, ffn_b2 + l * DD, TF, nullptr,
                NN, DD, FF, 0, stream);
    ln_res<<<NN / 8, 256, 0, stream>>>(TF, ln3_g + l * DD, ln3_b + l * DD, X, X, Xb, 0);
  }

  // ---- head ----
  launch_gemm(Xb, Wh1b, head_b1, nullptr, HNb, NN, DEMB, DD, 1, stream);
  launch_gemm(HNb, Wh2b, head_b2, out, nullptr, NN, TOUT, DEMB, 0, stream);
}
